// TransformerBlock_49134425866414
// MI455X (gfx1250) — matrix operand order repaired and checked by reading, not compiled, not run on hardware
//
#include <hip/hip_runtime.h>

// ---------------------------------------------------------------------------
// TransformerBlock for MI455X (gfx1250), wave32 + WMMA bf16 + async/TDM staging.
// Shapes: B=1, S=2048, D=4096, H=32, DH=128, FF=11008 (all tiles divide evenly).
// Compute-bound (~0.97 TFLOP vs ~0.8 GB weight traffic -> 34us @ 23.3 TB/s),
// so every matmul runs on v_wmma_f32_16x16x32_bf16 with f32 accumulation.
// ---------------------------------------------------------------------------

#define DEV __device__ __forceinline__

constexpr int S_  = 2048;
constexpr int D_  = 4096;
constexpr int H_  = 32;
constexpr int DH_ = 128;
constexpr int FF_ = 11008;

typedef __attribute__((ext_vector_type(16))) __bf16 v16bf;
typedef __attribute__((ext_vector_type(8)))  float  v8f;
typedef unsigned int v4u  __attribute__((ext_vector_type(4)));
typedef int          v8i_t __attribute__((ext_vector_type(8)));
typedef int          v4i_t __attribute__((ext_vector_type(4)));

union FragAB {           // 32 bytes: one WMMA bf16 A or B operand (8 VGPRs)
  v16bf v;
  uint4 q[2];
  unsigned short s[16];
};

// fp32 -> bf16 via native conversion (backend emits hw cvt, RNE)
DEV unsigned short f2bf(float f) {
  union { __bf16 b; unsigned short u; } x;
  x.b = (__bf16)f;
  return x.u;
}

// Generic pointer to LDS byte offset (flat LDS addresses carry offset in [31:0]).
DEV unsigned ldsoff(const void* p) { return (unsigned)(size_t)p; }

// Async global->LDS 16B copy (per-lane addresses). Tracked by ASYNCcnt.
DEV void async_ld_b128(unsigned dst_lds, const void* src_global) {
  asm volatile("global_load_async_to_lds_b128 %0, %1, off"
               :: "v"(dst_lds), "v"(src_global) : "memory");
}
DEV void wait_asynccnt0() { asm volatile("s_wait_asynccnt 0x0" ::: "memory"); }

DEV v8f wmma_bf16(const FragAB& a, const FragAB& b, v8f c) {
  return __builtin_amdgcn_wmma_f32_16x16x32_bf16(
      false, a.v, false, b.v, (short)0, c, false, false);
}

// ---------------------------------------------------------------------------
// Tensor Data Mover: 2-D bf16 tile load with LDS row padding (D# per ISA §8).
// This toolchain exposes the 6-arg builtin form:
//   (uint32x4 g0, int32x8 g1, int32x4 g2, int32x4 g3, int32x8 extra, i32 cpol)
// Falls back to per-lane async copies if the builtin is absent.
// ---------------------------------------------------------------------------
#if defined(__has_builtin)
#if __has_builtin(__builtin_amdgcn_tensor_load_to_lds) && \
    __has_builtin(__builtin_amdgcn_s_wait_tensorcnt)
#define HAVE_TDM 1
#endif
#endif
#ifndef HAVE_TDM
#define HAVE_TDM 0
#endif

#if HAVE_TDM
DEV void tdm_load_2d_bf16(unsigned lds_addr, const void* gaddr,
                          unsigned tdim0, unsigned tdim1,
                          unsigned tile0, unsigned tile1,
                          unsigned stride0,
                          unsigned pad_interval, unsigned pad_amount,
                          unsigned pad_en) {
  const unsigned long long ga = (unsigned long long)gaddr;
  v4u g0;
  g0[0] = 1u;                                            // count=1 (user D#)
  g0[1] = lds_addr;                                      // lds_addr [63:32]
  g0[2] = (unsigned)(ga & 0xFFFFFFFFu);                  // global_addr lo
  g0[3] = (unsigned)((ga >> 32) & 0x01FFFFFFu)           // global_addr [56:32]
        | (2u << 30);                                    // type=2 ("image")
  v8i_t g1;
  g1[0] = (int)((1u << 16)                               // data_size=1 (2B)
        | (pad_en << 20) | (pad_interval << 22) | (pad_amount << 25));
  g1[1] = (int)((tdim0 & 0xFFFFu) << 16);                // tensor_dim0 [79:48]
  g1[2] = (int)(((tdim0 >> 16) & 0xFFFFu) | ((tdim1 & 0xFFFFu) << 16));
  g1[3] = (int)(((tdim1 >> 16) & 0xFFFFu) | ((tile0 & 0xFFFFu) << 16));
  g1[4] = (int)(tile1 & 0xFFFFu);                        // tile_dim1 (tile_dim2=0)
  g1[5] = (int)stride0;                                  // tensor_dim0_stride lo
  g1[6] = 0;                                             // stride0 hi / stride1 lo
  g1[7] = 0;
  const v4i_t z4 = {0, 0, 0, 0};
  const v8i_t z8 = {0, 0, 0, 0, 0, 0, 0, 0};
  __builtin_amdgcn_tensor_load_to_lds(g0, g1, z4, z4, z8, 0);
}
#endif

// ---------------------------------------------------------------------------
// RMSNorm: one block per row of [S, D]; fp32 in, bf16 out.
// ---------------------------------------------------------------------------
__global__ __launch_bounds__(256)
void rmsnorm_bf16(const float* __restrict__ x, const float* __restrict__ w,
                  unsigned short* __restrict__ out) {
  const int s = blockIdx.x;
  const float* row = x + (size_t)s * D_;
  float acc = 0.f;
  for (int i = threadIdx.x; i < D_; i += 256) { float v = row[i]; acc += v * v; }
  for (int off = 16; off; off >>= 1) acc += __shfl_xor(acc, off, 32);
  __shared__ float red[8];
  __shared__ float rsc;
  const int lane = threadIdx.x & 31, wid = threadIdx.x >> 5;
  if (lane == 0) red[wid] = acc;
  __syncthreads();
  if (threadIdx.x == 0) {
    float t = 0.f;
    for (int i = 0; i < 8; ++i) t += red[i];
    rsc = rsqrtf(t / (float)D_ + 1e-5f);
  }
  __syncthreads();
  const float r = rsc;
  for (int i = threadIdx.x; i < D_; i += 256)
    out[(size_t)s * D_ + i] = f2bf(row[i] * r * w[i]);
}

// ---------------------------------------------------------------------------
// GEMM: C[M,N](f32) = A[M,K](bf16) x W[N,K](f32, converted inline)^T
//   RESID: C += resid    GLU: A tile = silu(Ag)*Au from fp32 inputs
// 128x128 block tile, 8 waves (4Mx2N), per-wave 32x64. K-step 32.
// Double-buffered LDS; A tile staged by async global->LDS copies, W tile
// staged through registers (needs fp32->bf16 conversion).
// ---------------------------------------------------------------------------
template <bool RESID, bool GLU>
__global__ __launch_bounds__(256)
void gemm_bf16_wT(float* __restrict__ C,
                  const unsigned short* __restrict__ A,   // bf16 [M,K] (!GLU)
                  const float* __restrict__ Ag,           // fp32 [M,K] (GLU)
                  const float* __restrict__ Au,           // fp32 [M,K] (GLU)
                  const float* __restrict__ W,            // fp32 [N,K]
                  int M, int N, int K,
                  const float* __restrict__ resid) {
  constexpr int LDT = 40;                         // 32 + 8 pad (bf16 units)
  __shared__ __align__(16) unsigned short As[2][128 * LDT];
  __shared__ __align__(16) unsigned short Bs[2][128 * LDT];

  const int tid  = threadIdx.x;
  const int lane = tid & 31, wid = tid >> 5;
  const int wm = wid & 3, wn = wid >> 2;
  const int bm = blockIdx.y * 128, bn = blockIdx.x * 128;
  const int lr = tid >> 1;                        // tile row 0..127
  const int lp = (tid & 1) * 16;                  // k offset 0 or 16
  const int fr = lane & 15, fh = lane >> 4;

  v8f acc[2][4];
#pragma unroll
  for (int i = 0; i < 2; ++i)
#pragma unroll
    for (int j = 0; j < 4; ++j)
#pragma unroll
      for (int e = 0; e < 8; ++e) acc[i][j][e] = 0.f;

  float4 wreg[4];
  float4 greg[4], ureg[4];

  auto issueA = [&](int buf, int kk) {
    if constexpr (!GLU) {                         // async byte copy to LDS
      const unsigned short* src = A + (size_t)(bm + lr) * K + kk + lp;
      async_ld_b128(ldsoff(&As[buf][lr * LDT + lp]), src);
      async_ld_b128(ldsoff(&As[buf][lr * LDT + lp + 8]), src + 8);
    } else {                                      // fp32 gate/up into regs
      const float* gs = Ag + (size_t)(bm + lr) * K + kk + lp;
      const float* us = Au + (size_t)(bm + lr) * K + kk + lp;
#pragma unroll
      for (int j = 0; j < 4; ++j) {
        greg[j] = *(const float4*)(gs + 4 * j);
        ureg[j] = *(const float4*)(us + 4 * j);
      }
    }
  };
  auto loadW = [&](int kk) {
    const float* ws2 = W + (size_t)(bn + lr) * K + kk + lp;
    __builtin_prefetch(ws2 + 32, 0, 1);           // global_prefetch next step
#pragma unroll
    for (int j = 0; j < 4; ++j) wreg[j] = *(const float4*)(ws2 + 4 * j);
  };
  auto storeW = [&](int buf) {
    alignas(16) unsigned short tmp[16];
#pragma unroll
    for (int j = 0; j < 4; ++j) {
      tmp[4 * j + 0] = f2bf(wreg[j].x);
      tmp[4 * j + 1] = f2bf(wreg[j].y);
      tmp[4 * j + 2] = f2bf(wreg[j].z);
      tmp[4 * j + 3] = f2bf(wreg[j].w);
    }
    *(uint4*)&Bs[buf][lr * LDT + lp]     = *(const uint4*)&tmp[0];
    *(uint4*)&Bs[buf][lr * LDT + lp + 8] = *(const uint4*)&tmp[8];
  };
  auto storeAglu = [&](int buf) {
    alignas(16) unsigned short tmp[16];
#pragma unroll
    for (int j = 0; j < 4; ++j) {
      const float4 gv = greg[j], uv = ureg[j];
      tmp[4 * j + 0] = f2bf((gv.x / (1.f + __expf(-gv.x))) * uv.x);
      tmp[4 * j + 1] = f2bf((gv.y / (1.f + __expf(-gv.y))) * uv.y);
      tmp[4 * j + 2] = f2bf((gv.z / (1.f + __expf(-gv.z))) * uv.z);
      tmp[4 * j + 3] = f2bf((gv.w / (1.f + __expf(-gv.w))) * uv.w);
    }
    *(uint4*)&As[buf][lr * LDT + lp]     = *(const uint4*)&tmp[0];
    *(uint4*)&As[buf][lr * LDT + lp + 8] = *(const uint4*)&tmp[8];
  };

  // prologue: stage tile 0
  issueA(0, 0);
  loadW(0);
  if constexpr (GLU) storeAglu(0);
  storeW(0);
  wait_asynccnt0();
  __syncthreads();

  const int nt = K / 32;
  for (int t = 0; t < nt; ++t) {
    const int cur = t & 1, nxt = cur ^ 1;
    if (t + 1 < nt) { issueA(nxt, (t + 1) * 32); loadW((t + 1) * 32); }

    FragAB fa[2], fb[4];
#pragma unroll
    for (int i = 0; i < 2; ++i) {
      const unsigned short* p = &As[cur][(wm * 32 + i * 16 + fr) * LDT + fh * 8];
      fa[i].q[0] = *(const uint4*)(p);
      fa[i].q[1] = *(const uint4*)(p + 16);
    }
#pragma unroll
    for (int j = 0; j < 4; ++j) {
      const unsigned short* p = &Bs[cur][(wn * 64 + j * 16 + fr) * LDT + fh * 8];
      fb[j].q[0] = *(const uint4*)(p);
      fb[j].q[1] = *(const uint4*)(p + 16);
    }
#pragma unroll
    for (int i = 0; i < 2; ++i)
#pragma unroll
      for (int j = 0; j < 4; ++j) acc[i][j] = wmma_bf16(fa[i], fb[j], acc[i][j]);

    if (t + 1 < nt) {
      storeW(nxt);
      if constexpr (GLU) storeAglu(nxt);
    }
    wait_asynccnt0();
    __syncthreads();
  }

  // epilogue: C/D layout (VGPR e = row, lane = col) -> global
#pragma unroll
  for (int i = 0; i < 2; ++i)
#pragma unroll
    for (int j = 0; j < 4; ++j) {
      const int col = bn + wn * 64 + j * 16 + fr;
#pragma unroll
      for (int e = 0; e < 8; ++e) {
        const int row = bm + wm * 32 + i * 16 + e + fh * 8;
        const size_t idx = (size_t)row * N + col;
        float v = acc[i][j][e];
        if constexpr (RESID) v += resid[idx];
        C[idx] = v;
      }
    }
}

// ---------------------------------------------------------------------------
// RoPE + repack: q,k fp32 [S,H,DH] -> bf16 [H,S,DH] (roped);
//                v fp32 [S,H,DH] -> bf16 [H,DH,S] (TRANSPOSED, so the flash
//                kernel's V tile is a straight async copy, B-frag contiguous).
// ---------------------------------------------------------------------------
__global__ __launch_bounds__(128)
void rope_qkv(const float* __restrict__ qf, const float* __restrict__ kf,
              const float* __restrict__ vf, const float* __restrict__ cs,
              const float* __restrict__ sn, unsigned short* __restrict__ qb,
              unsigned short* __restrict__ kb, unsigned short* __restrict__ vb) {
  const int b = blockIdx.x;
  const int s = b / H_, h = b % H_;
  const int t = threadIdx.x;
  const size_t src = ((size_t)s * H_ + h) * DH_;
  const size_t dst = ((size_t)h * S_ + s) * DH_;
  vb[((size_t)h * DH_ + t) * S_ + s] = f2bf(vf[src + t]);   // transposed store
  if (t < DH_ / 2) {
    const float c  = cs[(size_t)s * (DH_ / 2) + t];
    const float sv = sn[(size_t)s * (DH_ / 2) + t];
    const float qe = qf[src + 2 * t], qo = qf[src + 2 * t + 1];
    qb[dst + 2 * t]     = f2bf(qe * c - qo * sv);
    qb[dst + 2 * t + 1] = f2bf(qe * sv + qo * c);
    const float ke = kf[src + 2 * t], ko = kf[src + 2 * t + 1];
    kb[dst + 2 * t]     = f2bf(ke * c - ko * sv);
    kb[dst + 2 * t + 1] = f2bf(ke * sv + ko * c);
  }
}

// ---------------------------------------------------------------------------
// Flash attention (causal). Grid (S/128, H), 256 threads = 8 waves.
// Each wave owns 16 query rows; 32-key blocks; online softmax per wave.
// K tile via TDM (if available) else async copies; V tile via async copies.
// ---------------------------------------------------------------------------
__global__ __launch_bounds__(256)
void flash_attn(const unsigned short* __restrict__ Q,   // bf16 [H,S,DH]
                const unsigned short* __restrict__ Kg,  // bf16 [H,S,DH]
                const unsigned short* __restrict__ Vg,  // bf16 [H,DH,S]
                unsigned short* __restrict__ O) {       // bf16 [S,H*DH]
  constexpr int BQ = 128, BK = 32;
  constexpr int KLD = 136, VLD = 40, PLD = 40;
  __shared__ __align__(16) unsigned short Ks[BK * KLD];    // [key][dh]
  __shared__ __align__(16) unsigned short Vt[DH_ * VLD];   // [dh][key]
  __shared__ __align__(16) unsigned short Ps[8][16 * PLD]; // per-wave P

  const int h = blockIdx.y, qb = blockIdx.x;
  const int tid = threadIdx.x, lane = tid & 31, wid = tid >> 5;
  const int fr = lane & 15, fh = lane >> 4;
  const size_t headoff = (size_t)h * S_ * DH_;

  // Q fragments: 16 rows per wave, 4 k-chunks along DH, kept in VGPRs.
  const int qrow = qb * BQ + wid * 16 + fr;
  FragAB fq[4];
#pragma unroll
  for (int c = 0; c < 4; ++c) {
    const unsigned short* p = Q + headoff + (size_t)qrow * DH_ + c * 32 + fh * 8;
    fq[c].q[0] = *(const uint4*)(p);
    fq[c].q[1] = *(const uint4*)(p + 16);
  }

  v8f o[8];
#pragma unroll
  for (int j = 0; j < 8; ++j)
#pragma unroll
    for (int e = 0; e < 8; ++e) o[j][e] = 0.f;
  float m[8], l[8];
#pragma unroll
  for (int e = 0; e < 8; ++e) { m[e] = -3.0e38f; l[e] = 0.f; }

  const int nkb = (qb + 1) * (BQ / BK);
  const int lkey = tid >> 3, lpart = (tid & 7) * 16;   // K-tile loader
  const int vr = tid >> 1, vc = (tid & 1) * 16;        // V-tile loader

  for (int kb = 0; kb < nkb; ++kb) {
    const int k0 = kb * BK;
    // ---- stage K block [32 x 128] ----
#if HAVE_TDM
    if (wid == 0) {
      // tile: dim0=DH_ elems/row, dim1=BK rows; pad 4 DWORDs every 64 DWORDs
      // -> LDS row stride 136 bf16 (matches KLD).
      tdm_load_2d_bf16(ldsoff(Ks), Kg + headoff + (size_t)k0 * DH_,
                       DH_, S_, DH_, BK, DH_,
                       /*pad_interval=*/5, /*pad_amount=*/3, /*pad_en=*/1);
    }
#else
    {
      const unsigned short* src = Kg + headoff + (size_t)(k0 + lkey) * DH_ + lpart;
      async_ld_b128(ldsoff(&Ks[lkey * KLD + lpart]), src);
      async_ld_b128(ldsoff(&Ks[lkey * KLD + lpart + 8]), src + 8);
    }
#endif
    // ---- stage V block [128 dh x 32 keys] (already transposed in global) ----
    {
      const unsigned short* src = Vg + ((size_t)h * DH_ + vr) * S_ + k0 + vc;
      async_ld_b128(ldsoff(&Vt[vr * VLD + vc]), src);
      async_ld_b128(ldsoff(&Vt[vr * VLD + vc + 8]), src + 8);
    }
    wait_asynccnt0();
#if HAVE_TDM
    if (wid == 0) __builtin_amdgcn_s_wait_tensorcnt(0);
#endif
    __syncthreads();

    // ---- S = Q K^T (two 16x16 tiles: keys 0..15, 16..31) ----
    v8f s0, s1;
#pragma unroll
    for (int e = 0; e < 8; ++e) { s0[e] = 0.f; s1[e] = 0.f; }
#pragma unroll
    for (int c = 0; c < 4; ++c) {
      FragAB fb0, fb1;
      const unsigned short* p0 = &Ks[(fr)      * KLD + c * 32 + fh * 8];
      const unsigned short* p1 = &Ks[(16 + fr) * KLD + c * 32 + fh * 8];
      fb0.q[0] = *(const uint4*)(p0); fb0.q[1] = *(const uint4*)(p0 + 16);
      fb1.q[0] = *(const uint4*)(p1); fb1.q[1] = *(const uint4*)(p1 + 16);
      s0 = wmma_bf16(fq[c], fb0, s0);
      s1 = wmma_bf16(fq[c], fb1, s1);
    }

    // ---- scale + causal mask ----
    const float isq = 0.08838834764831845f;     // 1/sqrt(128)
    const int rbase = qb * BQ + wid * 16 + fh * 8;
#pragma unroll
    for (int e = 0; e < 8; ++e) {
      const int row = rbase + e;
      s0[e] = s0[e] * isq + ((k0 + fr)      > row ? -1e9f : 0.f);
      s1[e] = s1[e] * isq + ((k0 + 16 + fr) > row ? -1e9f : 0.f);
    }

    // ---- online softmax (stats per lane-half; width-16 reductions) ----
    float sc[8];
#pragma unroll
    for (int e = 0; e < 8; ++e) {
      float rx = fmaxf(s0[e], s1[e]);
      for (int off = 8; off; off >>= 1) rx = fmaxf(rx, __shfl_xor(rx, off, 16));
      const float mn = fmaxf(m[e], rx);
      sc[e] = __expf(m[e] - mn);
      const float p0 = __expf(s0[e] - mn);
      const float p1 = __expf(s1[e] - mn);
      float rs = p0 + p1;
      for (int off = 8; off; off >>= 1) rs += __shfl_xor(rs, off, 16);
      l[e] = l[e] * sc[e] + rs;
      m[e] = mn;
      s0[e] = p0; s1[e] = p1;
    }
#pragma unroll
    for (int j = 0; j < 8; ++j)
#pragma unroll
      for (int e = 0; e < 8; ++e) o[j][e] *= sc[e];

    // ---- restripe P: C layout -> LDS -> A layout (wave-local, DS in-order) ----
    unsigned short* pp = &Ps[wid][0];
#pragma unroll
    for (int e = 0; e < 8; ++e) {
      pp[(e + fh * 8) * PLD + fr]      = f2bf(s0[e]);
      pp[(e + fh * 8) * PLD + 16 + fr] = f2bf(s1[e]);
    }
    FragAB fp;
    const unsigned short* pa = &Ps[wid][fr * PLD + fh * 8];
    fp.q[0] = *(const uint4*)(pa);
    fp.q[1] = *(const uint4*)(pa + 16);

    // ---- O += P V ----
#pragma unroll
    for (int j = 0; j < 8; ++j) {
      FragAB fv;
      const unsigned short* pv = &Vt[(j * 16 + fr) * VLD + fh * 8];
      fv.q[0] = *(const uint4*)(pv);
      fv.q[1] = *(const uint4*)(pv + 16);
      o[j] = wmma_bf16(fp, fv, o[j]);
    }
    __syncthreads();
  }

  // ---- normalize + store bf16 [S, H*DH] ----
#pragma unroll
  for (int e = 0; e < 8; ++e) {
    const float inv = 1.f / l[e];
    const int row = qb * BQ + wid * 16 + fh * 8 + e;
#pragma unroll
    for (int j = 0; j < 8; ++j)
      O[(size_t)row * (H_ * DH_) + h * DH_ + j * 16 + fr] = f2bf(o[j][e] * inv);
  }
}

// ---------------------------------------------------------------------------
// Launch
// ---------------------------------------------------------------------------
extern "C" void kernel_launch(void* const* d_in, const int* in_sizes, int n_in,
                              void* d_out, int out_size, void* d_ws, size_t ws_size,
                              hipStream_t stream) {
  (void)in_sizes; (void)n_in; (void)out_size; (void)ws_size;
  const float* x    = (const float*)d_in[0];
  const float* fcos = (const float*)d_in[1];
  const float* fsin = (const float*)d_in[2];
  /* d_in[3] mask: causal mask applied analytically in flash_attn */
  const float* wq   = (const float*)d_in[4];
  const float* wk   = (const float*)d_in[5];
  const float* wv   = (const float*)d_in[6];
  const float* wo   = (const float*)d_in[7];
  const float* w1   = (const float*)d_in[8];
  const float* w2   = (const float*)d_in[9];
  const float* w3   = (const float*)d_in[10];
  const float* anw  = (const float*)d_in[11];
  const float* fnw  = (const float*)d_in[12];
  float* out = (float*)d_out;

  const size_t SD = (size_t)S_ * D_;
  const size_t SF = (size_t)S_ * FF_;

  char* p = (char*)d_ws;
  float*          h   = (float*)p;          p += SD * 4;   // residual stream
  unsigned short* hn  = (unsigned short*)p; p += SD * 2;
  unsigned short* xn  = (unsigned short*)p; p += SD * 2;
  float*          qf  = (float*)p;          p += SD * 4;
  float*          kf  = (float*)p;          p += SD * 4;
  float*          vf  = (float*)p;          p += SD * 4;
  unsigned short* qb  = (unsigned short*)p; p += SD * 2;
  unsigned short* kbb = (unsigned short*)p; p += SD * 2;
  unsigned short* vbb = (unsigned short*)p; p += SD * 2;
  unsigned short* att = (unsigned short*)p; p += SD * 2;
  float*          g   = qf;                 // alias (qf/kf/vf dead after rope)
  float*          u   = (float*)p;          p += SF * 4;

  const dim3 blk(256);
  const dim3 gdd(D_ / 128, S_ / 128);
  const dim3 gdf(FF_ / 128, S_ / 128);

  rmsnorm_bf16<<<S_, 256, 0, stream>>>(x, anw, xn);
  gemm_bf16_wT<false, false><<<gdd, blk, 0, stream>>>(qf, xn, nullptr, nullptr, wq, S_, D_, D_, nullptr);
  gemm_bf16_wT<false, false><<<gdd, blk, 0, stream>>>(kf, xn, nullptr, nullptr, wk, S_, D_, D_, nullptr);
  gemm_bf16_wT<false, false><<<gdd, blk, 0, stream>>>(vf, xn, nullptr, nullptr, wv, S_, D_, D_, nullptr);
  rope_qkv<<<S_ * H_, 128, 0, stream>>>(qf, kf, vf, fcos, fsin, qb, kbb, vbb);
  flash_attn<<<dim3(S_ / 128, H_), blk, 0, stream>>>(qb, kbb, vbb, att);
  gemm_bf16_wT<true, false><<<gdd, blk, 0, stream>>>(h, att, nullptr, nullptr, wo, S_, D_, D_, x);
  rmsnorm_bf16<<<S_, 256, 0, stream>>>(h, fnw, hn);
  gemm_bf16_wT<false, false><<<gdf, blk, 0, stream>>>(g, hn, nullptr, nullptr, w1, S_, FF_, D_, nullptr);
  gemm_bf16_wT<false, false><<<gdf, blk, 0, stream>>>(u, hn, nullptr, nullptr, w3, S_, FF_, D_, nullptr);
  gemm_bf16_wT<true, true><<<gdd, blk, 0, stream>>>(out, nullptr, g, u, w2, S_, D_, FF_, h);
}
